// Noise_Regressor_16587163697396
// MI455X (gfx1250) — matrix operand order repaired and checked by reading, not compiled, not run on hardware
//
#include <hip/hip_runtime.h>
#include <stdint.h>

typedef __attribute__((ext_vector_type(16))) __bf16 v16bf;
typedef __attribute__((ext_vector_type(8)))  float  v8f;

#define D_MODEL 1024
#define SEQ     4096
#define NCOL    864           // NUM_IMU * NUM_NOISE
#define NIMU    72
#define TSTEPS  300
#define OUTSZ   (NIMU * SEQ)  // 294912 elements per output tensor

// ---------- helpers ----------
__device__ __forceinline__ unsigned short f2bf(float f) {
  unsigned u = __float_as_uint(f);
  u += 0x7FFFu + ((u >> 16) & 1u);          // round-to-nearest-even
  return (unsigned short)(u >> 16);
}
__device__ __forceinline__ float bf2f(unsigned short h) {
  return __uint_as_float(((unsigned)h) << 16);
}
__device__ __forceinline__ float softplus_f(float x) {
  return fmaxf(x, 0.0f) + log1pf(expf(-fabsf(x)));
}

// ---------- 1) LayerNorm + bf16 hi/lo split of activations ----------
__global__ void __launch_bounds__(256)
ln_split_kernel(const float* __restrict__ x,
                const float* __restrict__ gamma,
                const float* __restrict__ beta,
                unsigned short* __restrict__ xh,
                unsigned short* __restrict__ xl) {
  __shared__ float red[256];
  const int row = blockIdx.x;
  const int tid = threadIdx.x;
  const float4 v = ((const float4*)(x + row * D_MODEL))[tid];

  float s  = v.x + v.y + v.z + v.w;
  float ss = v.x*v.x + v.y*v.y + v.z*v.z + v.w*v.w;

  red[tid] = s; __syncthreads();
  for (int o = 128; o > 0; o >>= 1) { if (tid < o) red[tid] += red[tid + o]; __syncthreads(); }
  const float mu = red[0] * (1.0f / D_MODEL);
  __syncthreads();
  red[tid] = ss; __syncthreads();
  for (int o = 128; o > 0; o >>= 1) { if (tid < o) red[tid] += red[tid + o]; __syncthreads(); }
  const float var = red[0] * (1.0f / D_MODEL) - mu * mu;
  const float rs  = rsqrtf(var + 1e-5f);

  const float xv[4] = {v.x, v.y, v.z, v.w};
  const int c0 = tid * 4;
#pragma unroll
  for (int j = 0; j < 4; ++j) {
    const int c = c0 + j;
    const float xn = (xv[j] - mu) * rs * gamma[c] + beta[c];
    const unsigned short h = f2bf(xn);
    xh[row * D_MODEL + c] = h;
    xl[row * D_MODEL + c] = f2bf(xn - bf2f(h));
  }
}

// ---------- 2) Transpose W (k-major -> n-major) + bf16 hi/lo split ----------
__global__ void __launch_bounds__(256)
prep_w_kernel(const float* __restrict__ W,
              unsigned short* __restrict__ wth,
              unsigned short* __restrict__ wtl) {
  const int idx = blockIdx.x * 256 + threadIdx.x;
  if (idx >= D_MODEL * NCOL) return;
  const int k = idx / NCOL;
  const int n = idx % NCOL;
  const float w = W[idx];
  const unsigned short h = f2bf(w);
  wth[n * D_MODEL + k] = h;
  wtl[n * D_MODEL + k] = f2bf(w - bf2f(h));
}

// ---------- 3) WMMA GEMM: p = xn @ W + b  (4096 x 864, K=1024) ----------
// One wave computes one 16x16 tile. bf16x2 split => 3 WMMAs per K-step.
__global__ void __launch_bounds__(256)
gemm_wmma_kernel(const unsigned short* __restrict__ xh,
                 const unsigned short* __restrict__ xl,
                 const unsigned short* __restrict__ wth,
                 const unsigned short* __restrict__ wtl,
                 const float* __restrict__ bias,
                 float* __restrict__ p) {
  const int wave = threadIdx.x >> 5;
  const int lane = threadIdx.x & 31;
  const int tile = blockIdx.x * 8 + wave;      // 13824 tiles total
  const int tm = tile & 255;                   // 256 M-tiles
  const int tn = tile >> 8;                    // 54  N-tiles
  const int l  = lane & 15;
  const bool hi = lane >= 16;

  const int m = tm * 16 + l;                   // A row handled by this lane
  const int n = tn * 16 + l;                   // B column handled by this lane
  const unsigned short* arow_h = xh  + m * D_MODEL;
  const unsigned short* arow_l = xl  + m * D_MODEL;
  const unsigned short* bcol_h = wth + n * D_MODEL;
  const unsigned short* bcol_l = wtl + n * D_MODEL;
  const int aoff = hi ? 16 : 0;                // A: lanes 16-31 hold K {16..31}
  const int boff = hi ? 16 : 0;                // B: lanes 16-31 hold K {16..31}

  v8f acc = {};
  for (int kb = 0; kb < D_MODEL; kb += 32) {
    union { v16bf v; uint4 q[2]; } ah, al, bh, bl;
    ah.q[0] = *(const uint4*)(arow_h + kb + aoff);
    ah.q[1] = *(const uint4*)(arow_h + kb + aoff + 8);
    al.q[0] = *(const uint4*)(arow_l + kb + aoff);
    al.q[1] = *(const uint4*)(arow_l + kb + aoff + 8);
    bh.q[0] = *(const uint4*)(bcol_h + kb + boff);
    bh.q[1] = *(const uint4*)(bcol_h + kb + boff + 8);
    bl.q[0] = *(const uint4*)(bcol_l + kb + boff);
    bl.q[1] = *(const uint4*)(bcol_l + kb + boff + 8);
    acc = __builtin_amdgcn_wmma_f32_16x16x32_bf16(false, ah.v, false, bh.v, (short)0, acc, false, false);
    acc = __builtin_amdgcn_wmma_f32_16x16x32_bf16(false, ah.v, false, bl.v, (short)0, acc, false, false);
    acc = __builtin_amdgcn_wmma_f32_16x16x32_bf16(false, al.v, false, bh.v, (short)0, acc, false, false);
  }

  const float bn = bias[n];
  const int m0 = tm * 16 + (hi ? 8 : 0);       // C/D: lanes 16-31 hold rows M=8..15
#pragma unroll
  for (int v = 0; v < 8; ++v) {
    p[(m0 + v) * NCOL + n] = acc[v] + bn;
  }
}

// ---------- 4) Parameter extraction + 4 direct outputs ----------
__global__ void __launch_bounds__(256)
params_kernel(const float* __restrict__ p,
              float* __restrict__ c,  float* __restrict__ hd,
              float* __restrict__ w1, float* __restrict__ ph,
              float* __restrict__ ct, float* __restrict__ hdt,
              float* __restrict__ w2, float* __restrict__ pht,
              float* __restrict__ out) {
  const int idx = blockIdx.x * 256 + threadIdx.x;      // idx = s*72 + i
  if (idx >= SEQ * NIMU) return;
  const int s = idx / NIMU;
  const int i = idx % NIMU;
  const float* pr = p + s * NCOL + i;

  const float p0 = pr[0*NIMU],  p1 = pr[1*NIMU],  p2  = pr[2*NIMU],  p3  = pr[3*NIMU];
  const float p4 = pr[4*NIMU],  p5 = pr[5*NIMU],  p6  = pr[6*NIMU],  p7  = pr[7*NIMU];
  const float p8 = pr[8*NIMU],  p9 = pr[9*NIMU],  p10 = pr[10*NIMU], p11 = pr[11*NIMU];

  const float d  = softplus_f(p1);
  const float k  = 0.25f * d * d + softplus_f(p0);
  const float dt = softplus_f(p3);
  const float kt = 0.25f * dt * dt + softplus_f(p2);

  const int o = i * SEQ + s;                   // imu-major for the gather kernel
  c[o]  = p4;  hd[o]  = 0.5f * d;
  w1[o] = 0.5f * sqrtf(fmaxf(4.0f * k  - d  * d,  0.0f));  ph[o]  = p6;
  ct[o] = p5;  hdt[o] = 0.5f * dt;
  w2[o] = 0.5f * sqrtf(fmaxf(4.0f * kt - dt * dt, 0.0f));  pht[o] = p7;

  out[1 * OUTSZ + o] = p8;                     // acc_base
  out[2 * OUTSZ + o] = softplus_f(p9);         // acc_std
  out[3 * OUTSZ + o] = p10;                    // gyro_base
  out[4 * OUTSZ + o] = softplus_f(p11);        // gyro_std
}

// ---------- 5) Kinematics: scatter-add rewritten as LDS-staged gather ----------
#define NSTAGE 560   // 256 + 299 rows needed, padded
__global__ void __launch_bounds__(256)
kin_kernel(const float* __restrict__ c,  const float* __restrict__ hd,
           const float* __restrict__ w1, const float* __restrict__ ph,
           const float* __restrict__ ct, const float* __restrict__ hdt,
           const float* __restrict__ w2, const float* __restrict__ pht,
           float* __restrict__ out) {
  __shared__ float sc[NSTAGE],  shd[NSTAGE],  sw1[NSTAGE], sph[NSTAGE];
  __shared__ float sct[NSTAGE], shdt[NSTAGE], sw2[NSTAGE], spht[NSTAGE];

  const int i    = blockIdx.y;                 // IMU
  const int j0   = blockIdx.x * 256;           // output chunk base
  const int base = i * SEQ;

  // stage params for s in [j0-299, j0+255]
  for (int li = threadIdx.x; li < 555; li += 256) {
    const int s = j0 - 299 + li;
    const bool ok = (s >= 0) && (s < SEQ);
    const int  so = base + (ok ? s : 0);
    const float m = ok ? 1.0f : 0.0f;
    sc[li]  = m * c[so];   shd[li]  = m * hd[so];
    sw1[li] = m * w1[so];  sph[li]  = m * ph[so];
    sct[li] = m * ct[so];  shdt[li] = m * hdt[so];
    sw2[li] = m * w2[so];  spht[li] = m * pht[so];
  }
  __syncthreads();

  const int j = j0 + (int)threadIdx.x;
  const int tmax = (j < TSTEPS - 1) ? j : (TSTEPS - 1);
  const int lbase = (int)threadIdx.x + 299;    // li = lbase - t  (s = j - t)
  float acc = 0.0f;
  for (int t = 0; t <= tmax; ++t) {
    const int li = lbase - t;
    const float tf = (float)t;
    const float lin = sc[li]  * __expf(-shd[li]  * tf) * __sinf(tf * sw1[li] + sph[li]);
    const float ang = sct[li] * __expf(-shdt[li] * tf) * __sinf(tf * sw2[li] + spht[li]);
    acc += lin + ang;
  }
  out[base + j] = acc;                         // kinematics at d_out offset 0
}

// ---------- launch ----------
extern "C" void kernel_launch(void* const* d_in, const int* in_sizes, int n_in,
                              void* d_out, int out_size, void* d_ws, size_t ws_size,
                              hipStream_t stream) {
  const float* hs    = (const float*)d_in[0];   // (1, 4096, 1024) fp32
  const float* gamma = (const float*)d_in[1];   // (1024,)
  const float* beta  = (const float*)d_in[2];   // (1024,)
  const float* W     = (const float*)d_in[3];   // (1024, 864)
  const float* bias  = (const float*)d_in[4];   // (864,)
  float* out = (float*)d_out;

  char* ws = (char*)d_ws;
  size_t off = 0;
  auto alloc = [&](size_t bytes) -> void* {
    void* ptr = ws + off;
    off = (off + bytes + 255) & ~(size_t)255;
    return ptr;
  };
  unsigned short* xh  = (unsigned short*)alloc((size_t)SEQ * D_MODEL * 2);
  unsigned short* xl  = (unsigned short*)alloc((size_t)SEQ * D_MODEL * 2);
  unsigned short* wth = (unsigned short*)alloc((size_t)NCOL * D_MODEL * 2);
  unsigned short* wtl = (unsigned short*)alloc((size_t)NCOL * D_MODEL * 2);
  float* p   = (float*)alloc((size_t)SEQ * NCOL * 4);
  float* pc  = (float*)alloc((size_t)OUTSZ * 4);
  float* phd = (float*)alloc((size_t)OUTSZ * 4);
  float* pw1 = (float*)alloc((size_t)OUTSZ * 4);
  float* pph = (float*)alloc((size_t)OUTSZ * 4);
  float* pct = (float*)alloc((size_t)OUTSZ * 4);
  float* phdt= (float*)alloc((size_t)OUTSZ * 4);
  float* pw2 = (float*)alloc((size_t)OUTSZ * 4);
  float* ppht= (float*)alloc((size_t)OUTSZ * 4);

  ln_split_kernel<<<SEQ, 256, 0, stream>>>(hs, gamma, beta, xh, xl);
  prep_w_kernel<<<(D_MODEL * NCOL + 255) / 256, 256, 0, stream>>>(W, wth, wtl);
  gemm_wmma_kernel<<<(256 * 54) / 8, 256, 0, stream>>>(xh, xl, wth, wtl, bias, p);
  params_kernel<<<(SEQ * NIMU) / 256, 256, 0, stream>>>(p, pc, phd, pw1, pph,
                                                        pct, phdt, pw2, ppht, out);
  kin_kernel<<<dim3(SEQ / 256, NIMU), 256, 0, stream>>>(pc, phd, pw1, pph,
                                                        pct, phdt, pw2, ppht, out);
}
